// AdaFGLModel_59502476919396
// MI455X (gfx1250) — matrix operand order repaired and checked by reading, not compiled, not run on hardware
//
#include <hip/hip_runtime.h>
#include <math.h>

typedef __attribute__((ext_vector_type(16))) _Float16 v16h;
typedef __attribute__((ext_vector_type(8)))  _Float16 v8h;
typedef __attribute__((ext_vector_type(8)))  float    v8f;

#define WMMA_F16(a, b, c) \
  __builtin_amdgcn_wmma_f32_16x16x32_f16(false, (a), false, (b), (short)0, (c), false, false)

constexpr int Nn    = 6144;
constexpr int FEAT  = 128;
constexpr int HID   = 256;
constexpr int OUT   = 64;
constexpr int IN_SM = 512;
constexpr int NT    = Nn / 16;             // 384 row/col tiles
constexpr int STAT_BLOCKS = (NT * NT) / 8; // 18432 blocks, 1 tile per wave

// ---------------- vectorized f16 fragment loaders (ISA 7.12.2 layouts) ----------------

__device__ inline v16h cat16(v8h lo, v8h hi) {
  v16h r;
#pragma unroll
  for (int j = 0; j < 8; ++j) { r[j] = lo[j]; r[8 + j] = hi[j]; }
  return r;
}

// A fragment 16x32: row M=(lane&15); lane<16 holds K {0..7,16..23}, lane>=16 {8..15,24..31}.
// Two contiguous 16B (b128) loads per lane.
__device__ inline v16h load_a_f16(const _Float16* __restrict__ A, int row0, int k0, int lda) {
  int lane = threadIdx.x & 31;
  const _Float16* p = A + (size_t)(row0 + (lane & 15)) * lda + k0 + ((lane >> 4) << 3);
  return cat16(*(const v8h*)p, *(const v8h*)(p + 16));
}

// B fragment 32x16 from K-contiguous (transposed) storage Bt[N][K]:
// lane<16 -> N=lane, K={k0..k0+7,k0+16..k0+23}; lane>=16 -> N=lane-16, K={k0+8..k0+15,k0+24..k0+31}.
// Two contiguous 16B loads per lane.
__device__ inline v16h load_b_f16t(const _Float16* __restrict__ Bt, int k0, int col0, int ldk) {
  int lane = threadIdx.x & 31;
  const _Float16* p = Bt + (size_t)(col0 + (lane & 15)) * ldk + k0 + ((lane >> 4) << 3);
  return cat16(*(const v8h*)p, *(const v8h*)(p + 16));
}

// A fragment from a 16x32 f16 tile in LDS (row-major, stride 32) — two ds b128 reads.
__device__ inline v16h load_a_lds(const _Float16* base) {
  int lane = threadIdx.x & 31;
  const _Float16* p = base + (lane & 15) * 32 + ((lane >> 4) << 3);
  return cat16(*(const v8h*)p, *(const v8h*)(p + 16));
}

// ---------------- pre-pass conversion kernels ----------------

__global__ __launch_bounds__(256) void convert_f16_kernel(
    const float* __restrict__ src, _Float16* __restrict__ dst, int n) {
  int i = blockIdx.x * blockDim.x + threadIdx.x;
  if (i < n) dst[i] = (_Float16)src[i];
}

// dst[c*rows + r] = src[r*cols + c]   (f32 [rows x cols] -> f16 [cols x rows])
__global__ __launch_bounds__(256) void transpose_convert_kernel(
    const float* __restrict__ src, _Float16* __restrict__ dst, int rows, int cols) {
  int i = blockIdx.x * blockDim.x + threadIdx.x;
  if (i >= rows * cols) return;
  int r = i / cols, c = i % cols;
  dst[(size_t)c * rows + r] = (_Float16)src[i];
}

// ---------------- GEMM + bias + optional PReLU (one 16x16 tile / wave) ----------------
// Xh: f16 row-major [M x K]; Wt: f16 transposed weights [Nc x K]; writes f32 and/or f16.

__global__ __launch_bounds__(256) void gemm_f16_kernel(
    const _Float16* __restrict__ Xh, const _Float16* __restrict__ Wt,
    const float* __restrict__ bias, float* __restrict__ Yf, _Float16* __restrict__ Yh,
    int K, int Nc, const float* __restrict__ slope_p) {
  int wave = blockIdx.x * 8 + (threadIdx.x >> 5);
  int lane = threadIdx.x & 31;
  int ntn = Nc >> 4;
  int tm = wave / ntn, tn = wave % ntn;
  v8f acc = {};
  for (int k = 0; k < K; k += 32) {
    v16h a = load_a_f16(Xh, tm * 16, k, K);
    v16h b = load_b_f16t(Wt, k, tn * 16, K);
    acc = WMMA_F16(a, b, acc);
  }
  int n  = tn * 16 + (lane & 15);
  int mb = (lane >> 4) << 3;
  float bv = bias[n];
  float slope = slope_p ? slope_p[0] : 0.f;
#pragma unroll
  for (int v = 0; v < 8; ++v) {
    float x = acc[v] + bv;
    if (slope_p) x = (x >= 0.f) ? x : slope * x;
    size_t idx = (size_t)(tm * 16 + mb + v) * Nc + n;
    if (Yf) Yf[idx] = x;
    if (Yh) Yh[idx] = (_Float16)x;
  }
}

// ---------------- Row softmax + diagonal scale: s (f16), rinv ----------------

__global__ __launch_bounds__(256) void softmax_diag_kernel(
    const float* __restrict__ femb, _Float16* __restrict__ Sh, float* __restrict__ rinv) {
  int i = blockIdx.x * blockDim.x + threadIdx.x;
  if (i >= Nn) return;
  const float* r = femb + (size_t)i * OUT;
  float mx = -3.4e38f;
  for (int c = 0; c < OUT; ++c) mx = fmaxf(mx, r[c]);
  float e[OUT];
  float sum = 0.f;
  for (int c = 0; c < OUT; ++c) { e[c] = expf(r[c] - mx); sum += e[c]; }
  float inv = 1.f / sum;
  float d = 0.f;
  for (int c = 0; c < OUT; ++c) {
    float v = e[c] * inv;
    Sh[(size_t)i * OUT + c] = (_Float16)v;
    d += v * v;                        // diagonal of s @ s.T
  }
  rinv[i] = 1.f / fmaxf(sqrtf(d), 1e-9f);
}

// ---------------- On-the-fly pass over N x N: mean & max of normalized re ----------------

__global__ __launch_bounds__(256) void stats_kernel(
    const _Float16* __restrict__ Sh, const float* __restrict__ rinv,
    float* __restrict__ psum, float* __restrict__ pmax) {
  __shared__ float ssum[256], smax[256];
  int wave = blockIdx.x * 8 + (threadIdx.x >> 5);
  int lane = threadIdx.x & 31;
  int ti = wave / NT, tj = wave % NT;
  v8f c = {};
  c = WMMA_F16(load_a_f16(Sh, ti * 16, 0,  OUT), load_b_f16t(Sh, 0,  tj * 16, OUT), c);
  c = WMMA_F16(load_a_f16(Sh, ti * 16, 32, OUT), load_b_f16t(Sh, 32, tj * 16, OUT), c);
  int colL = lane & 15, mb = (lane >> 4) << 3;
  int col = tj * 16 + colL;
  float rj = rinv[col];
  float ls = 0.f, lm = -3.4e38f;
#pragma unroll
  for (int v = 0; v < 8; ++v) {
    int row = ti * 16 + mb + v;
    float val = (row == col) ? 0.f : c[v] * rinv[row] * rj;  // diag zeroed
    ls += val;
    lm = fmaxf(lm, val);
  }
  ssum[threadIdx.x] = ls;
  smax[threadIdx.x] = lm;
  __syncthreads();
  for (int o = 128; o > 0; o >>= 1) {
    if ((int)threadIdx.x < o) {
      ssum[threadIdx.x] += ssum[threadIdx.x + o];
      smax[threadIdx.x] = fmaxf(smax[threadIdx.x], smax[threadIdx.x + o]);
    }
    __syncthreads();
  }
  if (threadIdx.x == 0) { psum[blockIdx.x] = ssum[0]; pmax[blockIdx.x] = smax[0]; }
}

__global__ __launch_bounds__(256) void finalize_stats_kernel(
    const float* __restrict__ psum, const float* __restrict__ pmax, float* __restrict__ stats) {
  __shared__ float ss[256], sm[256];
  int t = threadIdx.x;
  float s = 0.f, m = -3.4e38f;
  for (int i = t; i < STAT_BLOCKS; i += 256) { s += psum[i]; m = fmaxf(m, pmax[i]); }
  ss[t] = s; sm[t] = m;
  __syncthreads();
  for (int o = 128; o > 0; o >>= 1) {
    if (t < o) { ss[t] += ss[t + o]; sm[t] = fmaxf(sm[t], sm[t + o]); }
    __syncthreads();
  }
  if (t == 0) {
    float mean = ss[0] / ((float)Nn * (float)Nn);
    stats[0] = mean;            // m
    stats[1] = sm[0] - mean;    // pos_diff = max - mean
  }
}

// ---------------- Fused propagation: recompute lre tiles, prelu, tile @ femb ----------------
// fembT: f16 [OUT x Nn] (transposed feature_emb) so B fragments are K-contiguous.

__global__ __launch_bounds__(256) void propagate_kernel(
    const _Float16* __restrict__ Sh, const float* __restrict__ rinv,
    const float* __restrict__ femb, const _Float16* __restrict__ fembT,
    const float* __restrict__ stats, const float* __restrict__ slope_p,
    float* __restrict__ out) {
  __shared__ _Float16 apos[8][16 * 32];   // per-wave pos A-tile (16 rows x 32 packed cols)
  __shared__ _Float16 aneg[8][16 * 32];
  __shared__ float red[2][16][OUT];       // cross-wave accumulators (pos, neg)

  const int ti   = blockIdx.x;            // 16-row stripe
  const int wid  = threadIdx.x >> 5;
  const int lane = threadIdx.x & 31;
  const float m     = stats[0];
  const float pdiff = stats[1];
  const float a     = slope_p[0];

  for (int t = threadIdx.x; t < 2 * 16 * OUT; t += blockDim.x) (&red[0][0][0])[t] = 0.f;
  __syncthreads();

  // Stripe A fragments for s @ s.T are loop-invariant: hoist.
  const v16h sa0 = load_a_f16(Sh, ti * 16, 0,  OUT);
  const v16h sa1 = load_a_f16(Sh, ti * 16, 32, OUT);

  v8f accp0 = {}, accp1 = {}, accp2 = {}, accp3 = {};
  v8f accn0 = {}, accn1 = {}, accn2 = {}, accn3 = {};

  const int colL = lane & 15;
  const int mb   = (lane >> 4) << 3;

  for (int jp = wid; jp < NT / 2; jp += 8) {   // 192 column-tile pairs, 24 per wave
    int tj0 = jp * 2;
#pragma unroll
    for (int h = 0; h < 2; ++h) {
      int tj = tj0 + h;
      v8f c = {};
      c = WMMA_F16(sa0, load_b_f16t(Sh, 0,  tj * 16, OUT), c);
      c = WMMA_F16(sa1, load_b_f16t(Sh, 32, tj * 16, OUT), c);
      int col = tj * 16 + colL;
      float rj = rinv[col];
#pragma unroll
      for (int v = 0; v < 8; ++v) {
        int row = ti * 16 + mb + v;
        float re = (row == col) ? 0.f : c[v] * rinv[row] * rj;
        float x = re - m;
        float l = (x > 0.f) ? (x / pdiff) : (-((x + m) / m));
        if (row == col) l += 1.f;                              // add_diag (+eye)
        float pv = (l >= 0.f) ? l : a * l;                     // prelu(lre)
        float nl = -l;
        float nv = (nl >= 0.f) ? nl : a * nl;                  // prelu(-lre)
        apos[wid][(mb + v) * 32 + h * 16 + colL] = (_Float16)pv;
        aneg[wid][(mb + v) * 32 + h * 16 + colL] = (_Float16)nv;
      }
    }
    asm volatile("s_wait_dscnt 0x0" ::: "memory");  // LDS writes -> A-fragment reads
    v16h pa = load_a_lds(&apos[wid][0]);
    v16h na = load_a_lds(&aneg[wid][0]);
    int krow = tj0 * 16;                            // 32 femb rows in this K pack
    v16h b0 = load_b_f16t(fembT, krow,  0, Nn);
    v16h b1 = load_b_f16t(fembT, krow, 16, Nn);
    v16h b2 = load_b_f16t(fembT, krow, 32, Nn);
    v16h b3 = load_b_f16t(fembT, krow, 48, Nn);
    accp0 = WMMA_F16(pa, b0, accp0);  accn0 = WMMA_F16(na, b0, accn0);
    accp1 = WMMA_F16(pa, b1, accp1);  accn1 = WMMA_F16(na, b1, accn1);
    accp2 = WMMA_F16(pa, b2, accp2);  accn2 = WMMA_F16(na, b2, accn2);
    accp3 = WMMA_F16(pa, b3, accp3);  accn3 = WMMA_F16(na, b3, accn3);
  }

  // Cross-wave reduction via LDS float atomics (ds_add_f32).
#pragma unroll
  for (int v = 0; v < 8; ++v) {
    atomicAdd(&red[0][mb + v][ 0 + colL], accp0[v]);
    atomicAdd(&red[0][mb + v][16 + colL], accp1[v]);
    atomicAdd(&red[0][mb + v][32 + colL], accp2[v]);
    atomicAdd(&red[0][mb + v][48 + colL], accp3[v]);
    atomicAdd(&red[1][mb + v][ 0 + colL], accn0[v]);
    atomicAdd(&red[1][mb + v][16 + colL], accn1[v]);
    atomicAdd(&red[1][mb + v][32 + colL], accn2[v]);
    atomicAdd(&red[1][mb + v][48 + colL], accn3[v]);
  }
  __syncthreads();

  // Row softmax of pos/neg propagation + combine with femb.
  if (threadIdx.x < 16) {
    int r = threadIdx.x;
    int grow = ti * 16 + r;
    float mp = -3.4e38f, mn = -3.4e38f;
    for (int c = 0; c < OUT; ++c) {
      mp = fmaxf(mp, red[0][r][c]);
      mn = fmaxf(mn, red[1][r][c]);
    }
    float sp = 0.f, sn = 0.f;
    for (int c = 0; c < OUT; ++c) {
      float ep = expf(red[0][r][c] - mp); red[0][r][c] = ep; sp += ep;
      float en = expf(red[1][r][c] - mn); red[1][r][c] = en; sn += en;
    }
    float isp = 1.f / sp, isn = 1.f / sn;
    for (int c = 0; c < OUT; ++c)
      out[(size_t)grow * OUT + c] =
          0.5f * (red[0][r][c] * isp - red[1][r][c] * isn + femb[(size_t)grow * OUT + c]);
  }
}

// ---------------- Host-side orchestration ----------------

extern "C" void kernel_launch(void* const* d_in, const int* in_sizes, int n_in,
                              void* d_out, int out_size, void* d_ws, size_t ws_size,
                              hipStream_t stream) {
  (void)in_sizes; (void)n_in; (void)out_size; (void)ws_size;
  const float* ori = (const float*)d_in[0];
  const float* smo = (const float*)d_in[1];
  // d_in[2] processed_feature, d_in[3] universal_re unused (beta == 0)
  const float* Ws0 = (const float*)d_in[4];  const float* bs0 = (const float*)d_in[5];
  const float* Ws1 = (const float*)d_in[6];  const float* bs1 = (const float*)d_in[7];
  const float* Wl0 = (const float*)d_in[8];  const float* bl0 = (const float*)d_in[9];
  const float* Wl1 = (const float*)d_in[10]; const float* bl1 = (const float*)d_in[11];
  const float* Wh0 = (const float*)d_in[12]; const float* bh0 = (const float*)d_in[13];
  const float* Wh1 = (const float*)d_in[14]; const float* bh1 = (const float*)d_in[15];
  const float* amodel = (const float*)d_in[16];
  const float* ahete  = (const float*)d_in[17];

  float* out0 = (float*)d_out;            // local_ori_emb
  float* out1 = out0 + Nn * OUT;          // local_smooth_emb
  float* out2 = out1 + Nn * OUT;          // local_message_propagation

  // ---- workspace carve-out (256B aligned chunks) ----
  char* wsp = (char*)d_ws;
  auto carve = [&](size_t bytes) -> char* {
    char* p = wsp;
    wsp += (bytes + 255) & ~(size_t)255;
    return p;
  };
  _Float16* smo_h = (_Float16*)carve((size_t)Nn * IN_SM * 2);
  _Float16* ori_h = (_Float16*)carve((size_t)Nn * FEAT * 2);
  _Float16* H_h   = (_Float16*)carve((size_t)Nn * HID * 2);
  _Float16* Ws0t  = (_Float16*)carve((size_t)IN_SM * HID * 2);
  _Float16* Wl0t  = (_Float16*)carve((size_t)FEAT * HID * 2);
  _Float16* Wh0t  = (_Float16*)carve((size_t)IN_SM * HID * 2);
  _Float16* Ws1t  = (_Float16*)carve((size_t)HID * OUT * 2);
  _Float16* Wl1t  = (_Float16*)carve((size_t)HID * OUT * 2);
  _Float16* Wh1t  = (_Float16*)carve((size_t)HID * OUT * 2);
  float*    femb  = (float*)carve((size_t)Nn * OUT * 4);
  _Float16* fembT = (_Float16*)carve((size_t)OUT * Nn * 2);
  _Float16* Sh    = (_Float16*)carve((size_t)Nn * OUT * 2);
  float*    rinv  = (float*)carve((size_t)Nn * 4);
  float*    psum  = (float*)carve((size_t)STAT_BLOCKS * 4);
  float*    pmax  = (float*)carve((size_t)STAT_BLOCKS * 4);
  float*    stats = (float*)carve(2 * 4);

  auto cvt = [&](const float* s, _Float16* d, int n) {
    hipLaunchKernelGGL(convert_f16_kernel, dim3((n + 255) / 256), dim3(256), 0, stream, s, d, n);
  };
  auto tcvt = [&](const float* s, _Float16* d, int rows, int cols) {
    int n = rows * cols;
    hipLaunchKernelGGL(transpose_convert_kernel, dim3((n + 255) / 256), dim3(256), 0, stream,
                       s, d, rows, cols);
  };
  auto gemm = [&](const _Float16* X, const _Float16* Wt, const float* b,
                  float* Yf, _Float16* Yh, int K, int Nc, const float* slope) {
    int waves = (Nn / 16) * (Nc / 16);
    hipLaunchKernelGGL(gemm_f16_kernel, dim3(waves / 8), dim3(256), 0, stream,
                       X, Wt, b, Yf, Yh, K, Nc, slope);
  };

  // f32 -> f16 staging of activations and transposed weights
  cvt(smo, smo_h, Nn * IN_SM);
  cvt(ori, ori_h, Nn * FEAT);
  tcvt(Ws0, Ws0t, IN_SM, HID);
  tcvt(Wl0, Wl0t, FEAT,  HID);
  tcvt(Wh0, Wh0t, IN_SM, HID);
  tcvt(Ws1, Ws1t, HID, OUT);
  tcvt(Wl1, Wl1t, HID, OUT);
  tcvt(Wh1, Wh1t, HID, OUT);

  // Three MLP branches (hidden PReLU written as f16, final layer f32)
  gemm(smo_h, Ws0t, bs0, nullptr, H_h, IN_SM, HID, amodel);
  gemm(H_h,   Ws1t, bs1, out1, nullptr, HID, OUT, nullptr);   // local_smooth_emb
  gemm(ori_h, Wl0t, bl0, nullptr, H_h, FEAT, HID, amodel);
  gemm(H_h,   Wl1t, bl1, out0, nullptr, HID, OUT, nullptr);   // local_ori_emb
  gemm(smo_h, Wh0t, bh0, nullptr, H_h, IN_SM, HID, ahete);
  gemm(H_h,   Wh1t, bh1, femb, nullptr, HID, OUT, nullptr);   // feature_emb (f32)

  hipLaunchKernelGGL(softmax_diag_kernel, dim3(Nn / 256), dim3(256), 0, stream,
                     femb, Sh, rinv);
  tcvt(femb, fembT, Nn, OUT);   // f16 transposed feature_emb for K-contiguous B frags
  hipLaunchKernelGGL(stats_kernel, dim3(STAT_BLOCKS), dim3(256), 0, stream,
                     Sh, rinv, psum, pmax);
  hipLaunchKernelGGL(finalize_stats_kernel, dim3(1), dim3(256), 0, stream,
                     psum, pmax, stats);
  hipLaunchKernelGGL(propagate_kernel, dim3(NT), dim3(256), 0, stream,
                     Sh, rinv, femb, fembT, stats, ahete, out2);
}
